// ParallelMLayer_70635032150162
// MI455X (gfx1250) — matrix operand order repaired and checked by reading, not compiled, not run on hardware
//
#include <hip/hip_runtime.h>
#include <hip/hip_bf16.h>

#define D_DIM   2048
#define B_DIM   4
#define L_DIM   4096
#define M_DIM   (B_DIM * L_DIM)   /* 16384 rows */
#define CHUNKS  64
#define TCH     (L_DIM / CHUNKS)  /* 64 steps per chunk */
#define BASE_DT 0.1f
#define LN_EPS  1e-5f

typedef __attribute__((ext_vector_type(16))) __bf16 v16bf;
typedef __attribute__((ext_vector_type(8)))  float  v8f;

#define GLOBAL_AS __attribute__((address_space(1)))
#define LDS_AS    __attribute__((address_space(3)))

#if defined(__gfx1250__) && __has_builtin(__builtin_amdgcn_global_load_async_to_lds_b32) && \
    __has_builtin(__builtin_amdgcn_s_wait_asynccnt)
#define HAVE_ASYNC_LDS 1
#else
#define HAVE_ASYNC_LDS 0
#endif

// ---------- helpers ----------

__device__ __forceinline__ unsigned short f2bf(float f) {
    unsigned int u = __float_as_uint(f);
    u += 0x7FFFu + ((u >> 16) & 1u);          // round-to-nearest-even
    return (unsigned short)(u >> 16);
}

// A-fragment: contiguous K order, same as B: lane<16 -> K 0..15 ; lane>=16 -> K 16..31
__device__ __forceinline__ v16bf load_a_frag(const unsigned short* base, int koff) {
    uint4 lo = *(const uint4*)(base + koff);        // 8 bf16
    uint4 hi = *(const uint4*)(base + koff + 8);    // 8 bf16
    struct { uint4 a, b; } tmp{lo, hi};
    v16bf v;
    __builtin_memcpy(&v, &tmp, 32);
    return v;
}

// B-fragment: each lane holds 16 contiguous K values of its column
__device__ __forceinline__ v16bf load_b_frag(const unsigned short* base) {
    uint4 lo = *(const uint4*)(base);
    uint4 hi = *(const uint4*)(base + 8);
    struct { uint4 a, b; } tmp{lo, hi};
    v16bf v;
    __builtin_memcpy(&v, &tmp, 32);
    return v;
}

#define WMMA_BF16(a, b, c) \
    __builtin_amdgcn_wmma_f32_16x16x32_bf16(false, (a), false, (b), (short)0, (c), false, false)

// ---------- 1) LayerNorm (f32 in -> bf16 out) ----------

__global__ __launch_bounds__(256) void ln_kernel(const float* __restrict__ force,
                                                 const float* __restrict__ gamma,
                                                 const float* __restrict__ beta,
                                                 unsigned short* __restrict__ fb) {
    __shared__ float red[256];
    const int row = blockIdx.x;
    const int tid = threadIdx.x;
    const float* x = force + (size_t)row * D_DIM;

    float vals[8];
    float s = 0.f;
#pragma unroll
    for (int i = 0; i < 8; ++i) { vals[i] = x[tid + i * 256]; s += vals[i]; }
    red[tid] = s; __syncthreads();
    for (int off = 128; off > 0; off >>= 1) {
        if (tid < off) red[tid] += red[tid + off];
        __syncthreads();
    }
    const float mean = red[0] * (1.0f / D_DIM);
    __syncthreads();

    float sq = 0.f;
#pragma unroll
    for (int i = 0; i < 8; ++i) { float d = vals[i] - mean; sq += d * d; }
    red[tid] = sq; __syncthreads();
    for (int off = 128; off > 0; off >>= 1) {
        if (tid < off) red[tid] += red[tid + off];
        __syncthreads();
    }
    const float rstd = rsqrtf(red[0] * (1.0f / D_DIM) + LN_EPS);

    unsigned short* out = fb + (size_t)row * D_DIM;
#pragma unroll
    for (int i = 0; i < 8; ++i) {
        const int d = tid + i * 256;
        out[d] = f2bf((vals[i] - mean) * rstd * gamma[d] + beta[d]);
    }
}

// ---------- 2) weight transpose + bf16 convert (Wt[n*D+k] = W[k*D+n]) ----------
// Uses gfx1250 async global->LDS copies (ASYNCcnt) when the builtins exist.

__global__ __launch_bounds__(256) void wt_kernel(const float* __restrict__ WA,
                                                 const float* __restrict__ Wdt,
                                                 const float* __restrict__ WB,
                                                 unsigned short* __restrict__ TA,
                                                 unsigned short* __restrict__ Tdt,
                                                 unsigned short* __restrict__ TB) {
    __shared__ float tile[32][33];
    const float* W;
    unsigned short* T;
    if (blockIdx.z == 0)      { W = WA;  T = TA;  }
    else if (blockIdx.z == 1) { W = Wdt; T = Tdt; }
    else                      { W = WB;  T = TB;  }

    const int tx = threadIdx.x, ty = threadIdx.y;      // 32 x 8
    const int i0 = blockIdx.y * 32, j0 = blockIdx.x * 32;

#if HAVE_ASYNC_LDS
#pragma unroll
    for (int k = 0; k < 4; ++k) {
        const float* g = W + (size_t)(i0 + ty + k * 8) * D_DIM + (j0 + tx);
        __builtin_amdgcn_global_load_async_to_lds_b32(
            (GLOBAL_AS int*)g, (LDS_AS int*)&tile[ty + k * 8][tx], 0, 0);
    }
    __builtin_amdgcn_s_wait_asynccnt(0);   // this wave's async copies landed in LDS
    __syncthreads();                       // all waves' copies visible
#else
#pragma unroll
    for (int k = 0; k < 4; ++k)
        tile[ty + k * 8][tx] = W[(size_t)(i0 + ty + k * 8) * D_DIM + (j0 + tx)];
    __syncthreads();
#endif

#pragma unroll
    for (int k = 0; k < 4; ++k)
        T[(size_t)(j0 + ty + k * 8) * D_DIM + (i0 + tx)] = f2bf(tile[tx][ty + k * 8]);
}

// ---------- 3) fused GEMM + activation epilogue ----------
// z=0: sigmoid(f@W_A + b_A) -> outA
// z=1: softplus(f@W_dt + b_dt) * BASE_DT * dt_scales -> outdt
// z=2: (f@W_B + b_B) -> outBraw        (multiplied by dt in the scan)

__global__ __launch_bounds__(256) void gemm_kernel(const unsigned short* __restrict__ fb,
                                                   const unsigned short* __restrict__ TA,
                                                   const unsigned short* __restrict__ Tdt,
                                                   const unsigned short* __restrict__ TB,
                                                   const float* __restrict__ bA,
                                                   const float* __restrict__ bdt,
                                                   const float* __restrict__ bB,
                                                   const float* __restrict__ dt_scales,
                                                   float* __restrict__ outA,
                                                   float* __restrict__ outdt,
                                                   float* __restrict__ outB) {
    const int z = blockIdx.z;
    const unsigned short* Wt = (z == 0) ? TA : (z == 1) ? Tdt : TB;

    const int lane = threadIdx.x & 31;
    const int wave = threadIdx.x >> 5;
    const int row0 = blockIdx.y * 128 + wave * 16;     // 16 rows per wave
    const int col0 = blockIdx.x * 64;                  // 64 cols per wave
    const int m    = lane & 15;
    const int koffA = (lane >> 4) * 16;                // A-fragment K split
    const int koffB = (lane >> 4) * 16;                // B-fragment K split

    v8f c0 = {}, c1 = {}, c2 = {}, c3 = {};
    const unsigned short* arow  = fb + (size_t)(row0 + m) * D_DIM;
    const unsigned short* bcol0 = Wt + (size_t)(col0 + m) * D_DIM + koffB;
    const unsigned short* bcol1 = bcol0 + (size_t)16 * D_DIM;
    const unsigned short* bcol2 = bcol0 + (size_t)32 * D_DIM;
    const unsigned short* bcol3 = bcol0 + (size_t)48 * D_DIM;

    // software pipeline: fragments for step kk are loaded one iteration ahead,
    // so WMMAs only wait on the previous load group, not the current one.
    v16bf a  = load_a_frag(arow, koffA);
    v16bf w0 = load_b_frag(bcol0);
    v16bf w1 = load_b_frag(bcol1);
    v16bf w2 = load_b_frag(bcol2);
    v16bf w3 = load_b_frag(bcol3);

    for (int kk = 32; kk < D_DIM; kk += 32) {
        __builtin_prefetch(arow + kk + 512, 0, 3);     // near-cache prefetch ahead
        v16bf an  = load_a_frag(arow + kk, koffA);
        v16bf w0n = load_b_frag(bcol0 + kk);
        v16bf w1n = load_b_frag(bcol1 + kk);
        v16bf w2n = load_b_frag(bcol2 + kk);
        v16bf w3n = load_b_frag(bcol3 + kk);
        c0 = WMMA_BF16(a, w0, c0);
        c1 = WMMA_BF16(a, w1, c1);
        c2 = WMMA_BF16(a, w2, c2);
        c3 = WMMA_BF16(a, w3, c3);
        a = an; w0 = w0n; w1 = w1n; w2 = w2n; w3 = w3n;
    }
    c0 = WMMA_BF16(a, w0, c0);
    c1 = WMMA_BF16(a, w1, c1);
    c2 = WMMA_BF16(a, w2, c2);
    c3 = WMMA_BF16(a, w3, c3);

    // C layout: lane%16 = N, VGPR r = M (lanes 0-15) / M+8 (lanes 16-31)
    const int rbase = row0 + 8 * (lane >> 4);
    const int cbase = col0 + (lane & 15);
#pragma unroll
    for (int j = 0; j < 4; ++j) {
        const int col = cbase + j * 16;
        v8f c = (j == 0) ? c0 : (j == 1) ? c1 : (j == 2) ? c2 : c3;
        const float bias = (z == 0) ? bA[col] : (z == 1) ? bdt[col] : bB[col];
        const float scl  = (z == 1) ? dt_scales[col] * BASE_DT : 1.0f;
        float* dst = (z == 0) ? outA : (z == 1) ? outdt : outB;
#pragma unroll
        for (int r = 0; r < 8; ++r) {
            const float y = c[r] + bias;
            float o;
            if (z == 0)      o = 1.0f / (1.0f + __expf(-y));                           // sigmoid
            else if (z == 1) o = (fmaxf(y, 0.f) + log1pf(__expf(-fabsf(y)))) * scl;    // softplus*dt
            else             o = y;
            dst[(size_t)(rbase + r) * D_DIM + col] = o;
        }
    }
}

// ---------- 4) chunked associative scan ----------
// v_t = A_t v_{t-1} + Braw_t*dt_t ; x_t = sum v*dt

__global__ __launch_bounds__(256) void scan1_kernel(const float* __restrict__ Ag,
                                                    const float* __restrict__ dtg,
                                                    const float* __restrict__ Braw,
                                                    float* __restrict__ cA, float* __restrict__ cV,
                                                    float* __restrict__ cS1, float* __restrict__ cS2) {
    const int idx  = blockIdx.x * 256 + threadIdx.x;   // over B*CHUNKS*D
    const int d    = idx & (D_DIM - 1);
    const int rest = idx >> 11;
    const int ch   = rest & (CHUNKS - 1);
    const int b    = rest >> 6;
    const size_t base = ((size_t)b * L_DIM + (size_t)ch * TCH) * D_DIM + d;

    float aprod = 1.f, v = 0.f, s1 = 0.f, s2 = 0.f;
    for (int t = 0; t < TCH; ++t) {
        const size_t off = base + (size_t)t * D_DIM;
        const float A = Ag[off], dt = dtg[off];
        const float Bv = Braw[off] * dt;
        v = A * v + Bv;          // chunk-local v (v_init = 0)
        aprod *= A;              // prefix product of A
        s1 += aprod * dt;        // carries v_init contribution to x
        s2 += v * dt;            // local contribution to x
    }
    const size_t ci = (size_t)(b * CHUNKS + ch) * D_DIM + d;
    cA[ci] = aprod; cV[ci] = v; cS1[ci] = s1; cS2[ci] = s2;
}

__global__ __launch_bounds__(256) void scan2_kernel(const float* __restrict__ cA,
                                                    const float* __restrict__ cV,
                                                    const float* __restrict__ cS1,
                                                    const float* __restrict__ cS2,
                                                    float* __restrict__ vin,
                                                    float* __restrict__ xin) {
    const int idx = blockIdx.x * 256 + threadIdx.x;    // over B*D
    const int d = idx & (D_DIM - 1);
    const int b = idx >> 11;
    float v = 0.f, x = 0.f;
    for (int c = 0; c < CHUNKS; ++c) {
        const size_t ci = (size_t)(b * CHUNKS + c) * D_DIM + d;
        vin[ci] = v;
        xin[ci] = x;
        x += v * cS1[ci] + cS2[ci];
        v  = cA[ci] * v + cV[ci];
    }
}

__global__ __launch_bounds__(256) void scan3_kernel(const float* __restrict__ Ag,
                                                    const float* __restrict__ dtg,
                                                    const float* __restrict__ Braw,
                                                    const float* __restrict__ vin,
                                                    const float* __restrict__ xin,
                                                    float* __restrict__ xout,
                                                    float* __restrict__ vout) {
    const int idx  = blockIdx.x * 256 + threadIdx.x;   // over B*CHUNKS*D
    const int d    = idx & (D_DIM - 1);
    const int rest = idx >> 11;
    const int ch   = rest & (CHUNKS - 1);
    const int b    = rest >> 6;
    const size_t base = ((size_t)b * L_DIM + (size_t)ch * TCH) * D_DIM + d;
    const size_t ci = (size_t)(b * CHUNKS + ch) * D_DIM + d;

    float v = vin[ci], x = xin[ci];
    for (int t = 0; t < TCH; ++t) {
        const size_t off = base + (size_t)t * D_DIM;
        const float A = Ag[off], dt = dtg[off];
        v = A * v + Braw[off] * dt;
        x += v * dt;
        vout[off] = v;
        xout[off] = x;
    }
}

// ---------- host launch ----------

extern "C" void kernel_launch(void* const* d_in, const int* in_sizes, int n_in,
                              void* d_out, int out_size, void* d_ws, size_t ws_size,
                              hipStream_t stream) {
    const float* force     = (const float*)d_in[0];
    const float* gamma     = (const float*)d_in[1];
    const float* beta      = (const float*)d_in[2];
    const float* W_A       = (const float*)d_in[3];
    const float* b_A       = (const float*)d_in[4];
    const float* W_B       = (const float*)d_in[5];
    const float* b_B       = (const float*)d_in[6];
    const float* W_dt      = (const float*)d_in[7];
    const float* b_dt      = (const float*)d_in[8];
    const float* dt_scales = (const float*)d_in[9];
    (void)in_sizes; (void)n_in; (void)out_size; (void)ws_size;

    char* ws = (char*)d_ws;
    size_t off = 0;
    auto alloc = [&](size_t bytes) -> char* {
        char* p = ws + off;
        off += (bytes + 255) & ~(size_t)255;
        return p;
    };
    unsigned short* fb  = (unsigned short*)alloc((size_t)M_DIM * D_DIM * 2);
    unsigned short* TA  = (unsigned short*)alloc((size_t)D_DIM * D_DIM * 2);
    unsigned short* Tdt = (unsigned short*)alloc((size_t)D_DIM * D_DIM * 2);
    unsigned short* TB  = (unsigned short*)alloc((size_t)D_DIM * D_DIM * 2);
    float* wsA  = (float*)alloc((size_t)M_DIM * D_DIM * 4);
    float* wsdt = (float*)alloc((size_t)M_DIM * D_DIM * 4);
    float* wsBr = (float*)alloc((size_t)M_DIM * D_DIM * 4);
    const size_t csz = (size_t)B_DIM * CHUNKS * D_DIM * 4;
    float* cA  = (float*)alloc(csz);
    float* cV  = (float*)alloc(csz);
    float* cS1 = (float*)alloc(csz);
    float* cS2 = (float*)alloc(csz);
    float* vin = (float*)alloc(csz);
    float* xin = (float*)alloc(csz);

    float* xout = (float*)d_out;
    float* vout = xout + (size_t)M_DIM * D_DIM;

    // 1) layernorm -> bf16 activations
    ln_kernel<<<M_DIM, 256, 0, stream>>>(force, gamma, beta, fb);
    // 2) transpose + convert the three weight matrices
    wt_kernel<<<dim3(D_DIM / 32, D_DIM / 32, 3), dim3(32, 8), 0, stream>>>(
        W_A, W_dt, W_B, TA, Tdt, TB);
    // 3) three fused GEMMs with activation epilogues
    gemm_kernel<<<dim3(D_DIM / 64, M_DIM / 128, 3), 256, 0, stream>>>(
        fb, TA, Tdt, TB, b_A, b_dt, b_B, dt_scales, wsA, wsdt, wsBr);
    // 4) three-pass chunked scan
    scan1_kernel<<<(B_DIM * CHUNKS * D_DIM) / 256, 256, 0, stream>>>(
        wsA, wsdt, wsBr, cA, cV, cS1, cS2);
    scan2_kernel<<<(B_DIM * D_DIM) / 256, 256, 0, stream>>>(
        cA, cV, cS1, cS2, vin, xin);
    scan3_kernel<<<(B_DIM * CHUNKS * D_DIM) / 256, 256, 0, stream>>>(
        wsA, wsdt, wsBr, vin, xin, xout, vout);
}
